// SpeakingEncoder_45320494907627
// MI455X (gfx1250) — matrix operand order repaired and checked by reading, not compiled, not run on hardware
//
#include <hip/hip_runtime.h>

typedef __attribute__((ext_vector_type(16))) _Float16 v16h;
typedef __attribute__((ext_vector_type(8)))  _Float16 v8h;
typedef __attribute__((ext_vector_type(8)))  float    v8f;

#define NROWS   16384          // B*S
#define DDIM    512
#define KCODES  8192
#define SEQ     2048

#define WG_ROWS     128        // rows per workgroup
#define WG_THREADS  128        // 4 waves (wave32)
#define ITER_CODES  128        // codes per N-iteration per WG
#define NSPLIT      4          // code-range split across WGs (reduced via atomic min)
#define CODES_PER_WG (KCODES / NSPLIT)        // 2048
#define NITERS       (CODES_PER_WG / ITER_CODES) // 16
#define CHUNKS       (DDIM / 32)              // 16 k-chunks of 32

// LDS layout (halves), padded to avoid bank conflicts on ds_load_b128
#define A_STRIDE 520           // 512 + 8 halves pad  (row stride 1040B)
#define B_STRIDE 40            // 32 + 8 halves pad   (row stride 80B)
#define LDS_A_HALVES (WG_ROWS * A_STRIDE)     // 66560 halves = 133120 B
#define LDS_B_HALVES (ITER_CODES * B_STRIDE)  // 5120 halves  = 10240 B
#define LDS_BYTES (2 * LDS_A_HALVES + 2 * 2 * LDS_B_HALVES + WG_ROWS * 8) // 154624

// CDNA5 async global->LDS copy (bypasses VGPRs, tracked by ASYNCcnt).
// GVS mode: mem_addr = SGPR base + (i32 VGPR byte offset); LDS addr = wave LDS base + VGPR.
#define ASYNC_B128_TO_LDS(ldsOff, gOff, base)                                  \
  asm volatile("global_load_async_to_lds_b128 %0, %1, %2"                      \
               :: "v"(ldsOff), "v"(gOff), "s"(base) : "memory")
#define ASYNC_WAIT() asm volatile("s_wait_asynccnt 0x0" ::: "memory")

// monotone float -> uint map (orders like the float)
__device__ __forceinline__ unsigned int f32_key(float f) {
  unsigned int u = __float_as_uint(f);
  return (u & 0x80000000u) ? ~u : (u | 0x80000000u);
}

// ---------------- fp32 -> f16 convert (8 elems/thread) ----------------
__global__ void cvt_f16_kernel(const float* __restrict__ src,
                               _Float16* __restrict__ dst, int n8) {
  int i = blockIdx.x * blockDim.x + threadIdx.x;
  if (i >= n8) return;
  float4 a = ((const float4*)src)[2 * i + 0];
  float4 b = ((const float4*)src)[2 * i + 1];
  v8h h;
  h[0] = (_Float16)a.x; h[1] = (_Float16)a.y; h[2] = (_Float16)a.z; h[3] = (_Float16)a.w;
  h[4] = (_Float16)b.x; h[5] = (_Float16)b.y; h[6] = (_Float16)b.z; h[7] = (_Float16)b.w;
  ((v8h*)dst)[i] = h;
}

// ---------------- c2[k] = sum(codebook[k,:]^2), one wave32 per row ----------------
__global__ void c2_kernel(const float* __restrict__ cb, float* __restrict__ c2) {
  int w    = (blockIdx.x * blockDim.x + threadIdx.x) >> 5; // global wave id == code row
  int lane = threadIdx.x & 31;
  const float4* row = (const float4*)(cb + (size_t)w * DDIM);
  float s = 0.f;
#pragma unroll
  for (int it = 0; it < 4; ++it) {
    float4 v = row[lane + it * 32];
    s += v.x * v.x + v.y * v.y + v.z * v.z + v.w * v.w;
  }
#pragma unroll
  for (int off = 16; off > 0; off >>= 1) s += __shfl_down(s, off, 32);
  if (lane == 0) c2[w] = s;
}

// ---------------- init per-row packed (dist,idx) keys ----------------
__global__ void initkeys_kernel(unsigned long long* __restrict__ keys) {
  keys[blockIdx.x * blockDim.x + threadIdx.x] = 0xFFFFFFFFFFFFFFFFull;
}

// ---------------- main: WMMA distance + running argmin ----------------
__global__ __launch_bounds__(WG_THREADS, 1)
void distargmin_kernel(const _Float16* __restrict__ xh,
                       const _Float16* __restrict__ cbh,
                       const float* __restrict__ c2,
                       unsigned long long* __restrict__ rowkeys) {
  extern __shared__ __align__(16) char lds_raw[];
  _Float16* Ablk  = (_Float16*)lds_raw;
  _Float16* Bbuf0 = Ablk + LDS_A_HALVES;
  _Float16* Bbuf1 = Bbuf0 + LDS_B_HALVES;
  unsigned long long* red = (unsigned long long*)(Bbuf1 + LDS_B_HALVES);

  const int tid    = threadIdx.x;
  const int lane   = tid & 31;
  const int wave   = tid >> 5;
  const int wave_m = wave >> 1;      // 0..1 : which 64-row half
  const int wave_n = wave & 1;       // 0..1 : which 64-code half
  const int l16    = lane & 15;
  const int half   = lane >> 4;

  const int row0  = blockIdx.x * WG_ROWS;
  const int code0 = blockIdx.y * CODES_PER_WG;

  // LDS byte offsets (low 32 bits of flat shared address == LDS offset)
  const unsigned ldsA  = (unsigned)(size_t)(void*)Ablk;
  const unsigned ldsB0 = (unsigned)(size_t)(void*)Bbuf0;
  const unsigned ldsB1 = (unsigned)(size_t)(void*)Bbuf1;

  // ---- stage A block (128 rows x 512 halves) via async b128 copies ----
#pragma unroll 8
  for (int u = 0; u < (WG_ROWS * 64 / WG_THREADS); ++u) {       // 64 iters
    int idx = tid + u * WG_THREADS;
    int r   = idx >> 6;
    int cg  = idx & 63;
    unsigned gOff = (unsigned)((row0 + r) * (DDIM * 2) + cg * 16);
    unsigned lOff = ldsA + (unsigned)(r * (A_STRIDE * 2) + cg * 16);
    ASYNC_B128_TO_LDS(lOff, gOff, xh);
  }
  red[tid] = 0xFFFFFFFFFFFFFFFFull;

  float bestS[4][8];
  int   bestC[4][8];
#pragma unroll
  for (int mt = 0; mt < 4; ++mt)
#pragma unroll
    for (int r = 0; r < 8; ++r) { bestS[mt][r] = 3.4e38f; bestC[mt][r] = 0; }

  auto stageB = [&](int iter, int c, unsigned ldsDst) {
#pragma unroll
    for (int u = 0; u < (ITER_CODES * 4 / WG_THREADS); ++u) {   // 4 iters
      int idx = tid + u * WG_THREADS;                           // 0..511
      int cr  = idx >> 2;                                       // code row 0..127
      int sg  = idx & 3;                                        // 8-half segment
      unsigned gOff =
          (unsigned)(((code0 + iter * ITER_CODES + cr) * DDIM + c * 32 + sg * 8) * 2);
      unsigned lOff = ldsDst + (unsigned)(cr * (B_STRIDE * 2) + sg * 16);
      ASYNC_B128_TO_LDS(lOff, gOff, cbh);
    }
  };

  const v8f vzero = {};
  stageB(0, 0, ldsB0);   // prefetch first B chunk (buffer parity: global chunk g -> g&1)

  for (int iter = 0; iter < NITERS; ++iter) {
    v8f acc[4][4];
#pragma unroll
    for (int mt = 0; mt < 4; ++mt)
#pragma unroll
      for (int nt = 0; nt < 4; ++nt) acc[mt][nt] = vzero;

    for (int c = 0; c < CHUNKS; ++c) {
      ASYNC_WAIT();          // my async writes (<= current chunk) have landed in LDS
      __syncthreads();       // everyone's writes visible; prev buffer consumers done
      if (c + 1 < CHUNKS)            stageB(iter, c + 1, ((c + 1) & 1) ? ldsB1 : ldsB0);
      else if (iter + 1 < NITERS)    stageB(iter + 1, 0, ldsB0);
      const _Float16* Bb = (c & 1) ? Bbuf1 : Bbuf0;

      // A fragments: lane(0-15) rows M=l16, K = {0..7,16..23}; lanes 16-31 K += 8
      v16h afrag[4];
#pragma unroll
      for (int mt = 0; mt < 4; ++mt) {
        const _Float16* p = Ablk + (wave_m * 64 + mt * 16 + l16) * A_STRIDE + c * 32 + half * 8;
        v8h lo = *(const v8h*)p;
        v8h hi = *(const v8h*)(p + 16);
        v16h a;
#pragma unroll
        for (int i = 0; i < 8; ++i) { a[i] = lo[i]; a[i + 8] = hi[i]; }
        afrag[mt] = a;
      }
      // B fragments: lane holds column n=l16 (a codebook row), K = {0..7,16..23}; lanes 16-31 K += 8
      v16h bfrag[4];
#pragma unroll
      for (int nt = 0; nt < 4; ++nt) {
        const _Float16* p = Bb + (wave_n * 64 + nt * 16 + l16) * B_STRIDE + half * 8;
        v8h lo = *(const v8h*)p;
        v8h hi = *(const v8h*)(p + 16);
        v16h b;
#pragma unroll
        for (int i = 0; i < 8; ++i) { b[i] = lo[i]; b[i + 8] = hi[i]; }
        bfrag[nt] = b;
      }
#pragma unroll
      for (int mt = 0; mt < 4; ++mt)
#pragma unroll
        for (int nt = 0; nt < 4; ++nt)
          acc[mt][nt] = __builtin_amdgcn_wmma_f32_16x16x32_f16(
              false, afrag[mt], false, bfrag[nt], (short)0, acc[mt][nt], false, false);
    }

    // epilogue: score = c2[n] - 2*dot ; fold into running per-lane min
#pragma unroll
    for (int nt = 0; nt < 4; ++nt) {
      int n = code0 + iter * ITER_CODES + wave_n * 64 + nt * 16 + l16;
      float cc = c2[n];
#pragma unroll
      for (int mt = 0; mt < 4; ++mt)
#pragma unroll
        for (int r = 0; r < 8; ++r) {
          float s = __builtin_fmaf(-2.f, acc[mt][nt][r], cc);
          if (s < bestS[mt][r]) { bestS[mt][r] = s; bestC[mt][r] = n; }
        }
    }
  }

  // cross-lane / cross-wave reduction via packed u64 min (idx in low bits -> ties pick lowest idx)
#pragma unroll
  for (int mt = 0; mt < 4; ++mt)
#pragma unroll
    for (int r = 0; r < 8; ++r) {
      int rl = wave_m * 64 + mt * 16 + r + 8 * half;   // local row this (vgpr,lane) holds
      unsigned long long key =
          ((unsigned long long)f32_key(bestS[mt][r]) << 32) | (unsigned int)bestC[mt][r];
      atomicMin(&red[rl], key);
    }
  __syncthreads();
  atomicMin(&rowkeys[row0 + tid], red[tid]);           // cross-WG (NSPLIT) reduce
}

// ---------------- gather + sinusoidal positional encoding ----------------
__global__ void gather_pe_kernel(const float* __restrict__ cb,
                                 const unsigned long long* __restrict__ rowkeys,
                                 float* __restrict__ out) {
  int row = blockIdx.x;
  int s   = row & (SEQ - 1);
  unsigned int idx = (unsigned int)(rowkeys[row] & 0xFFFFFFFFull);
  const float* src = cb + (size_t)idx * DDIM;
  float* dst = out + (size_t)row * DDIM;
  const float coef = 0.017988946039015984f;            // ln(10000)/512
#pragma unroll
  for (int u = 0; u < 2; ++u) {
    int d = threadIdx.x + u * 256;
    float ang = (float)s * expf(-coef * (float)(d & ~1));
    float pe  = (d & 1) ? cosf(ang) : sinf(ang);
    dst[d] = src[d] + pe;
  }
}

extern "C" void kernel_launch(void* const* d_in, const int* in_sizes, int n_in,
                              void* d_out, int out_size, void* d_ws, size_t ws_size,
                              hipStream_t stream) {
  (void)in_sizes; (void)n_in; (void)out_size; (void)ws_size;
  const float* x  = (const float*)d_in[0];   // [8,2048,512] fp32
  const float* cb = (const float*)d_in[1];   // [8192,512] fp32
  float* out = (float*)d_out;

  char* ws = (char*)d_ws;
  _Float16* xh  = (_Float16*)ws;                                   // 16 MB
  _Float16* cbh = (_Float16*)(ws + 16777216);                      // 8 MB
  float*    c2  = (float*)(ws + 16777216 + 8388608);               // 32 KB
  unsigned long long* rowkeys =
      (unsigned long long*)(ws + 16777216 + 8388608 + 32768);      // 128 KB

  cvt_f16_kernel<<<(NROWS * DDIM / 8) / 256, 256, 0, stream>>>(x, xh, NROWS * DDIM / 8);
  cvt_f16_kernel<<<(KCODES * DDIM / 8) / 256, 256, 0, stream>>>(cb, cbh, KCODES * DDIM / 8);
  c2_kernel<<<KCODES / 8, 256, 0, stream>>>(cb, c2);
  initkeys_kernel<<<NROWS / 256, 256, 0, stream>>>(rowkeys);

  hipFuncSetAttribute((const void*)distargmin_kernel,
                      hipFuncAttributeMaxDynamicSharedMemorySize, LDS_BYTES);
  distargmin_kernel<<<dim3(NROWS / WG_ROWS, NSPLIT), WG_THREADS, LDS_BYTES, stream>>>(
      xh, cbh, c2, rowkeys);

  gather_pe_kernel<<<NROWS, 256, 0, stream>>>(cb, rowkeys, out);
}
